// CommitRankingModule_83227876262470
// MI455X (gfx1250) — matrix operand order repaired and checked by reading, not compiled, not run on hardware
//
#include <hip/hip_runtime.h>
#include <hip/hip_bf16.h>
#include <math.h>

#define H 256
#define NH 8
#define DH 32
#define PE_ROWS 100

typedef __attribute__((ext_vector_type(16))) __bf16 v16bf;
typedef __attribute__((ext_vector_type(8)))  float v8f;

__device__ __forceinline__ unsigned short f2bf(float f) {
  unsigned u = __float_as_uint(f);
  u += 0x7FFFu + ((u >> 16) & 1u);           // round-to-nearest-even
  return (unsigned short)(u >> 16);
}
__device__ __forceinline__ unsigned fkey(float f) {  // monotonic float->uint
  unsigned u = __float_as_uint(f);
  return (u & 0x80000000u) ? ~u : (u | 0x80000000u);
}
__device__ __forceinline__ float fdec(unsigned k) {
  return (k & 0x80000000u) ? __uint_as_float(k ^ 0x80000000u) : __uint_as_float(~k);
}
__device__ __forceinline__ float gelu_exact(float x) {
  return 0.5f * x * (1.0f + erff(x * 0.70710678118654752f));
}

// ---------------- utility ----------------
__global__ void k_fill_zero(float* p, int n) {
  int i = blockIdx.x * blockDim.x + threadIdx.x;
  int stride = gridDim.x * blockDim.x;
  for (; i < n; i += stride) p[i] = 0.0f;
}

// sinusoidal PE table [PE_ROWS, H]
__global__ void k_pe(float* pe) {
  int p = blockIdx.x;           // row
  int i = threadIdx.x;          // 0..127 pair index
  float div = expf((float)(2 * i) * (-9.210340371976184f / (float)H));
  float a = (float)p * div;
  pe[p * H + 2 * i]     = sinf(a);
  pe[p * H + 2 * i + 1] = cosf(a);
}

// fold queries into k-projection, TRANSPOSED: wq[h][m] = scale * sum_d q[h,d]*k_w[h*32+d, m]
__global__ void k_prep_wq(const float* __restrict__ q, const float* __restrict__ kw,
                          const float* __restrict__ kb, float* __restrict__ wq,
                          float* __restrict__ cq) {
  int m = threadIdx.x; // 256
  const float scale = 0.17677669529663687f; // 1/sqrt(32)
  for (int h = 0; h < NH; ++h) {
    float s = 0.f;
    for (int d = 0; d < DH; ++d) s += q[h * DH + d] * kw[(size_t)(h * DH + d) * H + m];
    wq[h * H + m] = s * scale;               // [h][m] layout: conflict-free LDS reads
  }
  if (m < NH) {
    float s = 0.f;
    for (int d = 0; d < DH; ++d) s += q[m * DH + d] * kb[m * DH + d];
    cq[m] = s * scale;
  }
}

// pass 1: scores[n,h] = (emb[n]+pe[ci[n]]) . wq[h,:] + cq[h]; segment max + counts
__global__ void k_scores(const float* __restrict__ emb, const int* __restrict__ ci,
                         const float* __restrict__ pe, const float* __restrict__ wq,
                         const float* __restrict__ cq, float* __restrict__ sc,
                         unsigned* __restrict__ smax, float* __restrict__ counts, int N) {
  __shared__ float    wq_l[NH * H];          // [h][feat] -> lane-stride-1 LDS reads
  __shared__ float    cq_l[NH];
  __shared__ unsigned smax_l[PE_ROWS * NH];
  __shared__ float    cnt_l[PE_ROWS];
  int t = threadIdx.x;
  for (int i = t; i < NH * H; i += 256) wq_l[i] = wq[i];
  if (t < NH) cq_l[t] = cq[t];
  for (int i = t; i < PE_ROWS * NH; i += 256) smax_l[i] = 0u;
  for (int i = t; i < PE_ROWS; i += 256) cnt_l[i] = 0.f;
  __syncthreads();
  int lane = t & 31, wave = t >> 5;
  for (int j = 0; j < 8; ++j) {          // 8 nodes per wave, 64 per block
    int n = blockIdx.x * 64 + wave * 8 + j;
    if (n >= N) break;
    int c = ci[n];
    float acc[NH];
#pragma unroll
    for (int h = 0; h < NH; ++h) acc[h] = 0.f;
    const float* er = emb + (size_t)n * H;
    const float* pr = pe + (size_t)c * H;
#pragma unroll
    for (int k = 0; k < 8; ++k) {
      int idx = lane + 32 * k;
      float f = er[idx] + pr[idx];
#pragma unroll
      for (int h = 0; h < NH; ++h) acc[h] += f * wq_l[h * H + idx];
    }
    float red[NH];
#pragma unroll
    for (int h = 0; h < NH; ++h) {
      float v = acc[h];
      for (int off = 16; off > 0; off >>= 1) v += __shfl_down(v, off, 32);
      red[h] = v;
    }
    if (lane == 0) {
#pragma unroll
      for (int h = 0; h < NH; ++h) {
        float s = red[h] + cq_l[h];
        sc[(size_t)n * NH + h] = s;
        atomicMax(&smax_l[c * NH + h], fkey(s));
      }
      atomicAdd(&cnt_l[c], 1.0f);
    }
  }
  __syncthreads();
  for (int i = t; i < PE_ROWS * NH; i += 256) if (smax_l[i]) atomicMax(&smax[i], smax_l[i]);
  for (int i = t; i < PE_ROWS; i += 256)     if (cnt_l[i] != 0.f) atomicAdd(&counts[i], cnt_l[i]);
}

// e = exp(score - smax[seg]); denom = segment sum (smax decoded once into LDS)
__global__ void k_esum(const int* __restrict__ ci, const unsigned* __restrict__ smax,
                       float* __restrict__ sc, float* __restrict__ denom, int total) {
  __shared__ float den_l[PE_ROWS * NH];
  __shared__ float mx_l[PE_ROWS * NH];
  int t = threadIdx.x;
  for (int i = t; i < PE_ROWS * NH; i += 256) { den_l[i] = 0.f; mx_l[i] = fdec(smax[i]); }
  __syncthreads();
  int base = blockIdx.x * 8192;
  int end = min(total, base + 8192);
  for (int i = base + t; i < end; i += 256) {
    int n = i >> 3, h = i & 7;
    int c = ci[n];
    float e = __expf(sc[i] - mx_l[c * NH + h]);
    sc[i] = e;
    atomicAdd(&den_l[c * NH + h], e);
  }
  __syncthreads();
  for (int i = t; i < PE_ROWS * NH; i += 256) if (den_l[i] != 0.f) atomicAdd(&denom[i], den_l[i]);
}

// pass 2: attn = e/denom (written out), A[c,h,:] += attn * x   (block = (commit, slice))
__global__ void k_attn_accum(const float* __restrict__ emb, const int* __restrict__ ci,
                             const float* __restrict__ pe, const float* __restrict__ e,
                             const float* __restrict__ denom, float* __restrict__ attn_out,
                             float* __restrict__ A, int N, int S) {
  __shared__ float pe_c[H];
  __shared__ float invd[NH];
  __shared__ int   list[256];
  __shared__ float attnv[256 * NH];
  __shared__ int   lcount;
  int c = blockIdx.x, s = blockIdx.y, t = threadIdx.x;
  int chunk = (N + S - 1) / S;
  int beg = s * chunk, end = min(N, beg + chunk);
  pe_c[t] = pe[(size_t)c * H + t];
  if (t < NH) { float d = denom[c * NH + t]; invd[t] = d > 0.f ? 1.f / d : 0.f; }
  float acc[NH];
#pragma unroll
  for (int h = 0; h < NH; ++h) acc[h] = 0.f;
  __syncthreads();
  for (int base = beg; base < end; base += 256) {
    if (t == 0) lcount = 0;
    __syncthreads();
    int idx = base + t;
    if (idx < end && ci[idx] == c) { int p = atomicAdd(&lcount, 1); list[p] = idx; }
    __syncthreads();
    int m = lcount;
    for (int j = t; j < m * NH; j += 256) {
      int n = list[j >> 3]; int h = j & 7;
      float a = e[(size_t)n * NH + h] * invd[h];
      attnv[j] = a;
      attn_out[(size_t)n * NH + h] = a;
    }
    __syncthreads();
    for (int j = 0; j < m; ++j) {
      int n = list[j];
      float xv = emb[(size_t)n * H + t] + pe_c[t];
#pragma unroll
      for (int h = 0; h < NH; ++h) acc[h] += attnv[j * NH + h] * xv;
    }
    __syncthreads();
  }
#pragma unroll
  for (int h = 0; h < NH; ++h) atomicAdd(&A[(size_t)(c * NH + h) * H + t], acc[h]);
}

// head_out = A . v_w^T + v_b ; pooled = head_out . pool_w^T + pool_b ; LN ; zero empty
__global__ void k_head_pool(const float* __restrict__ A, const float* __restrict__ vw,
                            const float* __restrict__ vb, const float* __restrict__ pw,
                            const float* __restrict__ pb, const float* __restrict__ g,
                            const float* __restrict__ beta, const float* __restrict__ counts,
                            float* __restrict__ y) {
  __shared__ float hold[H];
  __shared__ float red[H];
  int c = blockIdx.x, t = threadIdx.x;
  int h = t >> 5;
  const float* Ar = A + (size_t)(c * NH + h) * H;
  const float* wr = vw + (size_t)t * H;
  float dot = 0.f;
  for (int m = 0; m < H; ++m) dot += Ar[m] * wr[m];
  hold[t] = dot + vb[t];
  __syncthreads();
  float p = 0.f;
  const float* pr = pw + (size_t)t * H;
  for (int m = 0; m < H; ++m) p += hold[m] * pr[m];
  p += pb[t];
  red[t] = p; __syncthreads();
  for (int o = 128; o > 0; o >>= 1) { if (t < o) red[t] += red[t + o]; __syncthreads(); }
  float mean = red[0] / (float)H; __syncthreads();
  float dv = p - mean; red[t] = dv * dv; __syncthreads();
  for (int o = 128; o > 0; o >>= 1) { if (t < o) red[t] += red[t + o]; __syncthreads(); }
  float var = red[0] / (float)H;
  float val = dv * rsqrtf(var + 1e-5f) * g[t] + beta[t];
  if (counts[c] <= 0.f) val = 0.f;
  y[(size_t)c * H + t] = val;
}

// y = LN(yin + r)
__global__ void k_add_ln(const float* yin, const float* r,
                         const float* __restrict__ g, const float* __restrict__ b,
                         float* yout) {
  __shared__ float red[H];
  int c = blockIdx.x, t = threadIdx.x;
  float v = yin[(size_t)c * H + t] + r[(size_t)c * H + t];
  red[t] = v; __syncthreads();
  for (int o = 128; o > 0; o >>= 1) { if (t < o) red[t] += red[t + o]; __syncthreads(); }
  float mean = red[0] / (float)H; __syncthreads();
  float dv = v - mean; red[t] = dv * dv; __syncthreads();
  for (int o = 128; o > 0; o >>= 1) { if (t < o) red[t] += red[t + o]; __syncthreads(); }
  float var = red[0] / (float)H;
  yout[(size_t)c * H + t] = dv * rsqrtf(var + 1e-5f) * g[t] + b[t];
}

// D[M,Nc] = X[M,K] @ W[Nc,K]^T + bias, optional exact GELU — bf16 WMMA, 4 waves/block
__global__ void k_gemm_wmma(const float* __restrict__ X, const float* __restrict__ W,
                            const float* __restrict__ bias, float* __restrict__ D,
                            int M, int Nc, int K, int act) {
  int lane = threadIdx.x & 31;
  int wave = threadIdx.x >> 5;
  int half = lane >> 4;
  int l16  = lane & 15;
  int nt   = blockIdx.y * 4 + wave;
  int col0 = nt * 16;
  if (col0 >= Nc) return;
  int row_a = blockIdx.x * 16 + l16;
  int mc = min(row_a, M - 1);
  int n = col0 + l16;
  const float* Xr = X + (size_t)mc * K;
  const float* Wr = W + (size_t)n * K;
  v8f acc = {0.f, 0.f, 0.f, 0.f, 0.f, 0.f, 0.f, 0.f};
  for (int k0 = 0; k0 < K; k0 += 32) {
    union { v16bf v; unsigned short u[16]; } a, b;
    // A (16x32 bf16): lane(half,m): elems 0..15 -> k in [16h, 16h+16)
#pragma unroll
    for (int e = 0; e < 16; ++e) a.u[e] = f2bf(Xr[k0 + 16 * half + e]);
    // B (32x16 bf16): lane(half,n): elems 0..15 -> k in [16h, 16h+16)
#pragma unroll
    for (int e = 0; e < 16; ++e) b.u[e] = f2bf(Wr[k0 + 16 * half + e]);
    acc = __builtin_amdgcn_wmma_f32_16x16x32_bf16(false, a.v, false, b.v,
                                                  (short)0, acc, false, false);
  }
  float bv = bias ? bias[n] : 0.f;
#pragma unroll
  for (int r = 0; r < 8; ++r) {
    int row = blockIdx.x * 16 + 8 * half + r;   // D elem r -> row 8h+r, col n
    if (row < M) {
      float val = acc[r] + bv;
      if (act == 1) val = gelu_exact(val);
      D[(size_t)row * Nc + n] = val;
    }
  }
}

// transformer attention softmax: attw[h,q,:] over C keys
__global__ void k_attn_sm(const float* __restrict__ qkv, float* __restrict__ attw, int C) {
  __shared__ float qv[DH];
  __shared__ float srow[128];
  int h = blockIdx.x, q = blockIdx.y, t = threadIdx.x; // 128 threads, C<=128
  if (t < DH) qv[t] = qkv[(size_t)q * (3 * H) + h * DH + t];
  __syncthreads();
  const float scale = 0.17677669529663687f;
  for (int j = t; j < C; j += 128) {
    const float* kr = qkv + (size_t)j * (3 * H) + H + h * DH;
    float d = 0.f;
    for (int dd = 0; dd < DH; ++dd) d += qv[dd] * kr[dd];
    srow[j] = d * scale;
  }
  __syncthreads();
  float mx = -3.4e38f;
  for (int j = 0; j < C; ++j) mx = fmaxf(mx, srow[j]);
  __syncthreads();
  for (int j = t; j < C; j += 128) srow[j] = __expf(srow[j] - mx);
  __syncthreads();
  float sum = 0.f;
  for (int j = 0; j < C; ++j) sum += srow[j];
  float inv = 1.f / sum;
  for (int j = t; j < C; j += 128) attw[((size_t)h * C + q) * C + j] = srow[j] * inv;
}

// o[q, h*32+d] = sum_j attw[h,q,j] * v[j,h,d]
__global__ void k_attn_o(const float* __restrict__ attw, const float* __restrict__ qkv,
                         float* __restrict__ o, int C) {
  int q = blockIdx.x, t = threadIdx.x; // 256
  int h = t >> 5;
  const float* ar = attw + ((size_t)h * C + q) * C;
  float s = 0.f;
  for (int j = 0; j < C; ++j) s += ar[j] * qkv[(size_t)j * (3 * H) + 2 * H + t];
  o[(size_t)q * H + t] = s;
}

// out[c] = h1[c,:] . head2_w + head2_b
__global__ void k_final(const float* __restrict__ h1, const float* __restrict__ w2,
                        const float* __restrict__ b2, float* __restrict__ out, int C) {
  __shared__ float red[128];
  int c = blockIdx.x, t = threadIdx.x; // 128
  red[t] = h1[(size_t)c * (H / 2) + t] * w2[t];
  __syncthreads();
  for (int o = 64; o > 0; o >>= 1) { if (t < o) red[t] += red[t + o]; __syncthreads(); }
  if (t == 0) out[c] = red[0] + b2[0];
}

extern "C" void kernel_launch(void* const* d_in, const int* in_sizes, int n_in,
                              void* d_out, int out_size, void* d_ws, size_t ws_size,
                              hipStream_t stream) {
  (void)n_in; (void)ws_size;
  const float* emb        = (const float*)d_in[0];
  const float* cqueries   = (const float*)d_in[1];
  const float* k_w        = (const float*)d_in[2];
  const float* k_b        = (const float*)d_in[3];
  const float* v_w        = (const float*)d_in[4];
  const float* v_b        = (const float*)d_in[5];
  const float* pool_w     = (const float*)d_in[6];
  const float* pool_b     = (const float*)d_in[7];
  const float* pool_g     = (const float*)d_in[8];
  const float* pool_beta  = (const float*)d_in[9];
  const float* in_proj_w  = (const float*)d_in[10];
  const float* in_proj_b  = (const float*)d_in[11];
  const float* attn_out_w = (const float*)d_in[12];
  const float* attn_out_b = (const float*)d_in[13];
  const float* norm1_g    = (const float*)d_in[14];
  const float* norm1_b    = (const float*)d_in[15];
  const float* norm2_g    = (const float*)d_in[16];
  const float* norm2_b    = (const float*)d_in[17];
  const float* lin1_w     = (const float*)d_in[18];
  const float* lin1_b     = (const float*)d_in[19];
  const float* lin2_w     = (const float*)d_in[20];
  const float* lin2_b     = (const float*)d_in[21];
  const float* head1_w    = (const float*)d_in[22];
  const float* head1_b    = (const float*)d_in[23];
  const float* head2_w    = (const float*)d_in[24];
  const float* head2_b    = (const float*)d_in[25];
  const int*   ci         = (const int*)d_in[26];

  int N  = in_sizes[0] / H;            // 200000
  int C  = out_size - N * NH;          // 100
  int L  = in_sizes[10] / (3 * H * H); // 2
  int HF = 4 * H;

  float* out      = (float*)d_out;
  float* attn_out = out + C;

  float* w = (float*)d_ws;
  float* pe = w;  w += (size_t)PE_ROWS * H;
  float* wq = w;  w += (size_t)NH * H;
  float* cq = w;  w += NH;
  float* sc = w;  w += (size_t)N * NH;
  float* zstart = w;
  unsigned* smax = (unsigned*)w; w += PE_ROWS * NH;
  float* denom  = w; w += PE_ROWS * NH;
  float* counts = w; w += PE_ROWS;
  float* A      = w; w += (size_t)C * NH * H;
  int zcount = (int)(w - zstart);
  float* y    = w; w += (size_t)C * H;
  float* qkv  = w; w += (size_t)C * 3 * H;
  float* attw = w; w += (size_t)NH * C * C;
  float* o    = w; w += (size_t)C * H;
  float* t1   = w; w += (size_t)C * HF;
  float* t2   = w; w += (size_t)C * H;
  float* h1   = w; w += (size_t)C * (H / 2);

  k_fill_zero<<<256, 256, 0, stream>>>(zstart, zcount);
  k_pe<<<PE_ROWS, H / 2, 0, stream>>>(pe);
  k_prep_wq<<<1, H, 0, stream>>>(cqueries, k_w, k_b, wq, cq);
  k_scores<<<(N + 63) / 64, 256, 0, stream>>>(emb, ci, pe, wq, cq, sc, smax, counts, N);
  k_esum<<<(N * NH + 8191) / 8192, 256, 0, stream>>>(ci, smax, sc, denom, N * NH);
  k_attn_accum<<<dim3(C, 8), 256, 0, stream>>>(emb, ci, pe, sc, denom, attn_out, A, N, 8);
  k_head_pool<<<C, H, 0, stream>>>(A, v_w, v_b, pool_w, pool_b, pool_g, pool_beta, counts, y);

  for (int l = 0; l < L; ++l) {
    k_gemm_wmma<<<dim3((C + 15) / 16, (3 * H + 63) / 64), 128, 0, stream>>>(
        y, in_proj_w + (size_t)l * 3 * H * H, in_proj_b + (size_t)l * 3 * H, qkv, C, 3 * H, H, 0);
    k_attn_sm<<<dim3(NH, C), 128, 0, stream>>>(qkv, attw, C);
    k_attn_o<<<C, H, 0, stream>>>(attw, qkv, o, C);
    k_gemm_wmma<<<dim3((C + 15) / 16, (H + 63) / 64), 128, 0, stream>>>(
        o, attn_out_w + (size_t)l * H * H, attn_out_b + (size_t)l * H, t2, C, H, H, 0);
    k_add_ln<<<C, H, 0, stream>>>(y, t2, norm1_g + (size_t)l * H, norm1_b + (size_t)l * H, y);
    k_gemm_wmma<<<dim3((C + 15) / 16, (HF + 63) / 64), 128, 0, stream>>>(
        y, lin1_w + (size_t)l * HF * H, lin1_b + (size_t)l * HF, t1, C, HF, H, 1);
    k_gemm_wmma<<<dim3((C + 15) / 16, (H + 63) / 64), 128, 0, stream>>>(
        t1, lin2_w + (size_t)l * H * HF, lin2_b + (size_t)l * H, t2, C, H, HF, 0);
    k_add_ln<<<C, H, 0, stream>>>(y, t2, norm2_g + (size_t)l * H, norm2_b + (size_t)l * H, y);
  }
  k_gemm_wmma<<<dim3((C + 15) / 16, (H / 2 + 63) / 64), 128, 0, stream>>>(
      y, head1_w, head1_b, h1, C, H / 2, H, 1);
  k_final<<<C, 128, 0, stream>>>(h1, head2_w, head2_b, out, C);
}
